// BotleneckAdapter_51634096833251
// MI455X (gfx1250) — matrix operand order repaired and checked by reading, not compiled, not run on hardware
//
#include <hip/hip_runtime.h>
#include <hip/hip_bf16.h>
#include <math.h>

typedef __bf16 bf16_t;
typedef __attribute__((ext_vector_type(16))) __bf16 v16bf;
typedef __attribute__((ext_vector_type(8)))  float  v8f;

#define D_MODEL   2048
#define R_DIM     32
#define TILE_M    16
#define KCH       32
#define NCHUNKS   (D_MODEL / KCH)   // 64
#define NTILES    (D_MODEL / 16)    // 128
#define WAVES_PB  8
#define LN_EPS    1e-5f

// LDS layout (bytes):
//  s_wdn  : [64 kchunk][2 nhalf][32 lane][16 bf16]  = 131072
//  s_wup  : [128 ntile][32 lane][16 bf16]           = 131072
//  s_gamma: [2048 f32]                              = 8192
//  s_beta : [2048 f32]                              = 8192
//  s_bup  : [2048 f32]                              = 8192
//  s_g    : [8 wave][16 m][32 r] bf16               = 8192
#define OFF_WUP   131072
#define OFF_GAM   262144
#define OFF_BET   270336
#define OFF_BUP   278528
#define OFF_G     286720
#define SMEM_BYTES 294912

__global__ __launch_bounds__(256)
void bottleneck_adapter_wmma_kernel(const float* __restrict__ x,
                                    const float* __restrict__ gamma,
                                    const float* __restrict__ beta,
                                    const float* __restrict__ w_down,
                                    const float* __restrict__ b_down,
                                    const float* __restrict__ w_up,
                                    const float* __restrict__ b_up,
                                    float* __restrict__ out)
{
    extern __shared__ char smem[];
    bf16_t* s_wdn   = (bf16_t*)(smem);
    bf16_t* s_wup   = (bf16_t*)(smem + OFF_WUP);
    float*  s_gamma = (float*) (smem + OFF_GAM);
    float*  s_beta  = (float*) (smem + OFF_BET);
    float*  s_bup   = (float*) (smem + OFF_BUP);
    bf16_t* s_g     = (bf16_t*)(smem + OFF_G);

    const int tid  = threadIdx.x;
    const int wave = tid >> 5;
    const int lane = tid & 31;

    // ---- one-time per-block staging: LN params + bf16 weight fragments ----
    for (int i = tid; i < D_MODEL; i += 256) {
        s_gamma[i] = gamma[i];
        s_beta[i]  = beta[i];
        s_bup[i]   = b_up[i];
    }
    // w_down (2048x32, k-major) -> B-fragment layout for 16x16x32 bf16 WMMA:
// lane l holds (N = (l&15)+16*h, K = c*32 + (l>>4)*8 + j + (j>>3)*8), j=0..15
    for (int s = tid; s < NCHUNKS * 2 * 32; s += 256) {
        const int c  = s >> 6;
        const int h  = (s >> 5) & 1;
        const int l  = s & 31;
        const int n  = (l & 15) + 16 * h;
        const int k0 = c * KCH + ((l >> 4) << 3);
        bf16_t* dst = s_wdn + (size_t)s * 16;
#pragma unroll
        for (int j = 0; j < 16; ++j)
            dst[j] = (bf16_t)w_down[(size_t)(k0 + j + ((j >> 3) << 3)) * R_DIM + n];
    }
    // w_up (32x2048, r-major) -> B-fragment layout, one 32x16 fragment per N-tile
    for (int s = tid; s < NTILES * 32; s += 256) {
        const int t  = s >> 5;
        const int l  = s & 31;
        const int n  = t * 16 + (l & 15);
        const int k0 = (l >> 4) << 3;
        bf16_t* dst = s_wup + (size_t)s * 16;
#pragma unroll
        for (int j = 0; j < 16; ++j)
            dst[j] = (bf16_t)w_up[(size_t)(k0 + j + ((j >> 3) << 3)) * D_MODEL + n];
    }
    __syncthreads();

    const int tile = blockIdx.x * WAVES_PB + wave;
    const int row0 = tile * TILE_M;
    const int mrow = lane & 15;     // row within tile this lane serves (A/C layout)
    const int half = lane >> 4;     // K/N half selector
    const size_t rowbase = (size_t)(row0 + mrow) * D_MODEL;

    // ---- pass 1: LayerNorm statistics (2 lanes per row, float4 stream) ----
    float s1 = 0.f, s2 = 0.f;
    {
        const float4* xr = (const float4*)(x + rowbase + half * (D_MODEL / 2));
#pragma unroll 4
        for (int i = 0; i < (D_MODEL / 2) / 4; ++i) {
            const float4 v = xr[i];
            s1 += v.x + v.y + v.z + v.w;
            s2 += v.x * v.x + v.y * v.y + v.z * v.z + v.w * v.w;
        }
    }
    s1 += __shfl_xor(s1, 16, 32);
    s2 += __shfl_xor(s2, 16, 32);
    const float mu   = s1 * (1.0f / D_MODEL);
    const float var  = s2 * (1.0f / D_MODEL) - mu * mu;
    const float rstd = rsqrtf(var + LN_EPS);

    auto lnf = [&](float v, int k) -> bf16_t {
        return (bf16_t)(((v - mu) * rstd) * s_gamma[k] + s_beta[k]);
    };

    // ---- pass 2: fused LN + down-projection via bf16 WMMA ----
    // A-fragment: lane<16 -> M=lane, K in {0..7,16..23}; lane>=16 -> K in {8..15,24..31}
    v8f acc0 = {};
    v8f acc1 = {};
    for (int c = 0; c < NCHUNKS; ++c) {
        const int k0 = c * KCH + half * 8;
        const float* xp = x + rowbase + k0;
        __builtin_prefetch(xp + 4 * KCH, 0, 0);   // global_prefetch next chunks
        const float4 xa = *(const float4*)(xp);
        const float4 xb = *(const float4*)(xp + 4);
        const float4 xc = *(const float4*)(xp + 16);
        const float4 xd = *(const float4*)(xp + 20);
        v16bf a;
        a[0]  = lnf(xa.x, k0 + 0);  a[1]  = lnf(xa.y, k0 + 1);
        a[2]  = lnf(xa.z, k0 + 2);  a[3]  = lnf(xa.w, k0 + 3);
        a[4]  = lnf(xb.x, k0 + 4);  a[5]  = lnf(xb.y, k0 + 5);
        a[6]  = lnf(xb.z, k0 + 6);  a[7]  = lnf(xb.w, k0 + 7);
        a[8]  = lnf(xc.x, k0 + 16); a[9]  = lnf(xc.y, k0 + 17);
        a[10] = lnf(xc.z, k0 + 18); a[11] = lnf(xc.w, k0 + 19);
        a[12] = lnf(xd.x, k0 + 20); a[13] = lnf(xd.y, k0 + 21);
        a[14] = lnf(xd.z, k0 + 22); a[15] = lnf(xd.w, k0 + 23);

        const v16bf b0 = *(const v16bf*)(s_wdn + ((size_t)(c * 2 + 0) * 32 + lane) * 16);
        const v16bf b1 = *(const v16bf*)(s_wdn + ((size_t)(c * 2 + 1) * 32 + lane) * 16);
        acc0 = __builtin_amdgcn_wmma_f32_16x16x32_bf16(false, a, false, b0,
                                                       (short)0, acc0, false, false);
        acc1 = __builtin_amdgcn_wmma_f32_16x16x32_bf16(false, a, false, b1,
                                                       (short)0, acc1, false, false);
    }

    // ---- bias + exact GELU (fp32), then re-swizzle 16x32 activations via LDS ----
    // C layout: lane l holds (M = v + 8*(l>>4), N = l&15) in acc[v]
    const float bd0 = b_down[lane & 15];
    const float bd1 = b_down[16 + (lane & 15)];
    bf16_t* gw = s_g + wave * (TILE_M * R_DIM);
#pragma unroll
    for (int v = 0; v < 8; ++v) {
        const int m = v + 8 * half;
        float z0 = acc0[v] + bd0;
        float z1 = acc1[v] + bd1;
        z0 = 0.5f * z0 * (1.0f + erff(z0 * 0.70710678118654752f));
        z1 = 0.5f * z1 * (1.0f + erff(z1 * 0.70710678118654752f));
        gw[m * R_DIM + (lane & 15)]      = (bf16_t)z0;
        gw[m * R_DIM + 16 + (lane & 15)] = (bf16_t)z1;
    }
    asm volatile("s_wait_dscnt 0" ::: "memory");   // same-wave LDS RAW before gather

    v16bf au;
    {
        const bf16_t* gr = gw + mrow * R_DIM;
        const int kb = half * 8;
#pragma unroll
        for (int j = 0; j < 8; ++j) {
            au[j]     = gr[kb + j];
            au[8 + j] = gr[kb + 16 + j];
        }
    }

    // ---- pass 3: up-projection (K=32, one WMMA per 16-col tile) + residual ----
    for (int t = 0; t < NTILES; ++t) {
        const v16bf b = *(const v16bf*)(s_wup + ((size_t)t * 32 + lane) * 16);
        v8f c = {};
        c = __builtin_amdgcn_wmma_f32_16x16x32_bf16(false, au, false, b,
                                                    (short)0, c, false, false);
        const int col = t * 16 + (lane & 15);
        const float bup = s_bup[col];
        const size_t base = (size_t)(row0 + 8 * half) * D_MODEL + col;
#pragma unroll
        for (int v = 0; v < 8; ++v) {
            const size_t idx = base + (size_t)v * D_MODEL;
            out[idx] = x[idx] + c[v] + bup;
        }
    }
}

extern "C" void kernel_launch(void* const* d_in, const int* in_sizes, int n_in,
                              void* d_out, int out_size, void* d_ws, size_t ws_size,
                              hipStream_t stream) {
    const float* x      = (const float*)d_in[0];
    const float* gamma  = (const float*)d_in[1];
    const float* beta   = (const float*)d_in[2];
    const float* w_down = (const float*)d_in[3];
    const float* b_down = (const float*)d_in[4];
    const float* w_up   = (const float*)d_in[5];
    const float* b_up   = (const float*)d_in[6];
    float* out = (float*)d_out;

    const int tokens = in_sizes[0] / D_MODEL;            // 16384
    const int tiles  = tokens / TILE_M;                  // 1024
    const int blocks = tiles / WAVES_PB;                 // 128

    (void)hipFuncSetAttribute((const void*)bottleneck_adapter_wmma_kernel,
                              hipFuncAttributeMaxDynamicSharedMemorySize,
                              SMEM_BYTES);

    bottleneck_adapter_wmma_kernel<<<blocks, 256, SMEM_BYTES, stream>>>(
        x, gamma, beta, w_down, b_down, w_up, b_up, out);
}
